// mlp_phonon_53961969107549
// MI455X (gfx1250) — matrix operand order repaired and checked by reading, not compiled, not run on hardware
//
#include <hip/hip_runtime.h>

// ---------------------------------------------------------------------------
// MI455X (gfx1250) fused pipeline for the phonon-DOS MLP.
// All GEMMs: V_WMMA_F32_16X16X32_F16 (f16 in, f32 accumulate).
// LDS tiles are stored in WMMA *fragment-major* layout (ISA 7.12.2): every
// fragment access is a contiguous 32B ds_load_b128/ds_store_b128 pair.
// K-chunks are double-buffered with register prefetch: one barrier/iteration,
// global->LDS staging overlapped with WMMA issue; prefetch chains across
// GEMM stages so the pipeline never drains.
// Edge branch of the reference is dead code (ea is deleted) and is skipped.
// ---------------------------------------------------------------------------

typedef _Float16 h16;
typedef __attribute__((ext_vector_type(16))) _Float16 v16h;
typedef __attribute__((ext_vector_type(8)))  float    v8f;

#define HD      256          // hidden width
#define FA      118          // atom-feature width
#define NNODES  65536
#define NGRAPH  1024
#define NE      51
#define LN_EPS  1e-5f
#define CH      512          // h16 elements per fragment chunk (32 lanes * 16)

#define WMMA_F16(a, b, c) \
  __builtin_amdgcn_wmma_f32_16x16x32_f16(false, (a), false, (b), (short)0, (c), false, false)

__device__ __forceinline__ v8f zero8() {
  v8f z;
#pragma unroll
  for (int i = 0; i < 8; ++i) z[i] = 0.f;
  return z;
}

__device__ __forceinline__ v16h zeroh() {
  v16h z;
#pragma unroll
  for (int i = 0; i < 16; ++i) z[i] = (h16)0.f;
  return z;
}

// Read one lane's fragment: 32 contiguous bytes -> 2x ds_load_b128.
__device__ __forceinline__ v16h read_frag(const h16* __restrict__ chunk, int lane) {
  return *(const v16h*)(chunk + lane * 16);
}

// ---- B chunk staging (32 x HD weight rows, row stride HD) ------------------
// Fragment-major B: 16 tiles x 32 lanes x 16 h16 (8192 h16 = 16KB).
// element (k,c) -> slot ((c>>4)*32 + (c&15) + 16*((k>>3)&1)), elem (k&7) + 8*(k>>4).
struct BReg { v16h r0, r1; };

__device__ __forceinline__ BReg load_b(const float* __restrict__ W, int k0,
                                       int kmax, int tid) {
  BReg out;
#pragma unroll
  for (int s = 0; s < 2; ++s) {
    int slot = tid + s * 256;
    int c  = ((slot >> 5) << 4) + (slot & 15);
    int hi = (slot >> 4) & 1;
    v16h v;
#pragma unroll
    for (int e = 0; e < 16; ++e) {
      int k = k0 + (e & 7) + 8 * hi + 16 * (e >> 3);
      v[e] = (h16)((k < kmax) ? W[(size_t)k * HD + c] : 0.f);
    }
    if (s == 0) out.r0 = v; else out.r1 = v;
  }
  return out;
}

__device__ __forceinline__ void store_b(h16* __restrict__ sB, int tid,
                                        const BReg& b) {
  *(v16h*)(sB + tid * 16) = b.r0;
  *(v16h*)(sB + (tid + 256) * 16) = b.r1;
}

// ---- A chunk staging (32 rows x 32 k, from an f32 row-major matrix) --------
// A-frag: elem e of lane (m,hi) holds k = (e&7) + 8*hi + 16*(e>>3).
// Valid for tid < 64 (slot = mtile*32 + lane).
__device__ __forceinline__ v16h load_a_rows(const float* __restrict__ P, int ld,
                                            int row0, int k0, int tid) {
  int mtile = tid >> 5, slane = tid & 31;
  int r = row0 + mtile * 16 + (slane & 15);
  int ahi = slane >> 4;
  v16h v;
#pragma unroll
  for (int e = 0; e < 16; ++e) {
    int k = k0 + (e & 7) + 8 * ahi + 16 * (e >> 3);
    v[e] = (h16)P[(size_t)r * ld + k];
  }
  return v;
}

// C-tile -> A-frag forward mapping (epilogues writing next-GEMM activations):
//   col c = nj*64 + t*16 + nn, row m = mi*16 + vv + 8*hic:
//   kc = nj*2 + (t>>1); lane' = (m&15) + 16*(nn>>3); elem = (nn&7) + 8*(t&1).

// ---------------------------------------------------------------------------
__global__ void k_zero(float* __restrict__ p, int n) {
  int i = blockIdx.x * blockDim.x + threadIdx.x;
  int stride = gridDim.x * blockDim.x;
  for (; i < n; i += stride) p[i] = 0.f;
}

// ---------------------------------------------------------------------------
// Node encoder (2-layer MLP with PReLU) fused with per-graph scatter-sum.
// grid.x = NNODES/32, grid.y in {0:z, 1:x}. Block = 256 threads (8 waves);
// wave (mi,nj) owns rows mi*16..+15, cols nj*64..+63 of the 32x256 tile.
__global__ __launch_bounds__(256) void k_node_pool(
    const float* __restrict__ x, const float* __restrict__ z,
    const int* __restrict__ batch,
    const float* __restrict__ w1, const float* __restrict__ b1,
    const float* __restrict__ pa1,
    const float* __restrict__ w2, const float* __restrict__ b2,
    float* __restrict__ pooled /* [NGRAPH][2*HD] */) {
  __shared__ __align__(32) h16 sA[2 * 4 * CH];   // frag-major input, 8KB
  __shared__ __align__(32) h16 sB[2][16 * CH];   // double-buffered, 32KB
  __shared__ __align__(32) h16 sH[2 * 8 * CH];   // frag-major hidden acts, 16KB
  __shared__ int sBatch[32];

  const int tid  = threadIdx.x;
  const int lane = tid & 31;
  const int wv   = tid >> 5;
  const int mi   = wv >> 2;   // 0..1
  const int nj   = wv & 3;    // 0..3
  const int hic  = (lane >> 4) & 1;
  const int nn   = lane & 15;
  const int row0 = blockIdx.x * 32;
  const int which = blockIdx.y;               // 0 = z branch, 1 = x branch
  const float* src = which ? x : z;
  const int colofs = which ? HD : 0;
  const float alpha = pa1[0];

  if (tid < 32) sBatch[tid] = batch[row0 + tid];

  // Stage full A (K padded 118->128) fragment-major: one 32B slot per thread.
  {
    int slane = tid & 31;
    int kc    = (tid >> 5) & 3;
    int mtile = tid >> 7;
    int r  = row0 + mtile * 16 + (slane & 15);
    int ahi = slane >> 4;
    v16h v;
#pragma unroll
    for (int e = 0; e < 16; ++e) {
      int k = kc * 32 + (e & 7) + 8 * ahi + 16 * (e >> 3);
      v[e] = (h16)((k < FA) ? src[(size_t)r * FA + k] : 0.f);
    }
    *(v16h*)(sA + tid * 16) = v;
  }

  BReg pb = load_b(w1, 0, FA, tid);
  store_b(sB[0], tid, pb);
  __syncthreads();

  // ---- GEMM1: h1 = prelu(X @ W1 + b1), 4 K-chunks, double-buffered ----
  v8f acc[4];
#pragma unroll
  for (int t = 0; t < 4; ++t) acc[t] = zero8();

#pragma unroll
  for (int kc = 0; kc < 4; ++kc) {
    const h16* cur = sB[kc & 1];
    h16* nxt = sB[(kc + 1) & 1];
    // prefetch: next W1 chunk, or W2 chunk 0 on the last iteration
    pb = (kc < 3) ? load_b(w1, (kc + 1) * 32, FA, tid)
                  : load_b(w2, 0, HD, tid);
    v16h a = read_frag(sA + (mi * 4 + kc) * CH, lane);
#pragma unroll
    for (int t = 0; t < 4; ++t) {
      v16h b = read_frag(cur + (nj * 4 + t) * CH, lane);
      acc[t] = WMMA_F16(a, b, acc[t]);
    }
    store_b(nxt, tid, pb);
    __syncthreads();
  }

  // Epilogue -> fragment-major sH (becomes A of GEMM2).
#pragma unroll
  for (int t = 0; t < 4; ++t) {
    int c = nj * 64 + t * 16 + nn;
    float bb = b1[c];
    int kcA  = nj * 2 + (t >> 1);
    int elem = (nn & 7) + 8 * (t & 1);
#pragma unroll
    for (int vv = 0; vv < 8; ++vv) {
      int m15 = vv + 8 * hic;
      float hv = acc[t][vv] + bb;
      hv = (hv >= 0.f) ? hv : alpha * hv;
      sH[((mi * 8 + kcA) * 32 + m15 + 16 * (nn >> 3)) * 16 + elem] = (h16)hv;
    }
  }
  __syncthreads();  // sH visible before GEMM2 reads (chunk0 of W2 already in sB[0])

  // ---- GEMM2: h2 = h1 @ W2 + b2, 8 K-chunks, scatter-add into pooled ----
  v8f acc2[4];
#pragma unroll
  for (int t = 0; t < 4; ++t) acc2[t] = zero8();

#pragma unroll
  for (int kc = 0; kc < 8; ++kc) {
    const h16* cur = sB[kc & 1];
    bool more = (kc < 7);
    if (more) pb = load_b(w2, (kc + 1) * 32, HD, tid);
    v16h a = read_frag(sH + (mi * 8 + kc) * CH, lane);
#pragma unroll
    for (int t = 0; t < 4; ++t) {
      v16h b = read_frag(cur + (nj * 4 + t) * CH, lane);
      acc2[t] = WMMA_F16(a, b, acc2[t]);
    }
    if (more) {
      store_b(sB[(kc + 1) & 1], tid, pb);
      __syncthreads();
    }
  }

#pragma unroll
  for (int t = 0; t < 4; ++t) {
    int c = nj * 64 + t * 16 + nn;
    float bb = b2[c];
#pragma unroll
    for (int vv = 0; vv < 8; ++vv) {
      int m = mi * 16 + vv + (hic ? 8 : 0);
      int gph = sBatch[m];
      atomicAdd(&pooled[(size_t)gph * (2 * HD) + colofs + c], acc2[t][vv] + bb);
    }
  }
}

// ---------------------------------------------------------------------------
// graph = concat(zp, xp) @ dec_w + dec_b.  grid = NGRAPH/32 blocks.
__global__ __launch_bounds__(256) void k_decoder(
    const float* __restrict__ pooled, const float* __restrict__ dw,
    const float* __restrict__ db, float* __restrict__ graph) {
  __shared__ __align__(32) h16 sA[2][2 * CH];    // double-buffered A chunk, 4KB
  __shared__ __align__(32) h16 sB[2][16 * CH];   // 32KB

  const int tid  = threadIdx.x;
  const int lane = tid & 31;
  const int wv   = tid >> 5;
  const int mi   = wv >> 2;
  const int nj   = wv & 3;
  const int hic  = (lane >> 4) & 1;
  const int nn   = lane & 15;
  const int row0 = blockIdx.x * 32;

  v8f acc[4];
#pragma unroll
  for (int t = 0; t < 4; ++t) acc[t] = zero8();

  BReg pb = load_b(dw, 0, 2 * HD, tid);
  v16h pa = zeroh();
  if (tid < 64) pa = load_a_rows(pooled, 2 * HD, row0, 0, tid);
  store_b(sB[0], tid, pb);
  if (tid < 64) *(v16h*)(sA[0] + tid * 16) = pa;
  __syncthreads();

  for (int kc = 0; kc < 16; ++kc) {
    const h16* curB = sB[kc & 1];
    const h16* curA = sA[kc & 1];
    bool more = (kc < 15);
    if (more) {
      pb = load_b(dw, (kc + 1) * 32, 2 * HD, tid);
      if (tid < 64) pa = load_a_rows(pooled, 2 * HD, row0, (kc + 1) * 32, tid);
    }
    v16h a = read_frag(curA + mi * CH, lane);
#pragma unroll
    for (int t = 0; t < 4; ++t) {
      v16h b = read_frag(curB + (nj * 4 + t) * CH, lane);
      acc[t] = WMMA_F16(a, b, acc[t]);
    }
    if (more) {
      store_b(sB[(kc + 1) & 1], tid, pb);
      if (tid < 64) *(v16h*)(sA[(kc + 1) & 1] + tid * 16) = pa;
      __syncthreads();
    }
  }

#pragma unroll
  for (int t = 0; t < 4; ++t) {
    int c = nj * 64 + t * 16 + nn;
    float bb = db[c];
#pragma unroll
    for (int vv = 0; vv < 8; ++vv) {
      int m = mi * 16 + vv + (hic ? 8 : 0);
      graph[(size_t)(row0 + m) * HD + c] = acc[t][vv] + bb;
    }
  }
}

// ---------------------------------------------------------------------------
// Head: feat=[emb(e)|graph(g)] @ w0 + b0 -> LayerNorm -> PReLU ->
// 3x (h @ w + b -> PReLU) -> h @ wl + bl -> dos[g][e].
// Rows r = e*1024 + g, 32 rows per block, grid = 51*1024/32 = 1632.
__global__ __launch_bounds__(256) void k_head(
    const float* __restrict__ emb, const float* __restrict__ graph,
    const float* __restrict__ w0, const float* __restrict__ b0,
    const float* __restrict__ lng, const float* __restrict__ lnb,
    const float* __restrict__ pa0,
    const float* __restrict__ w_l0, const float* __restrict__ b_l0,
    const float* __restrict__ a_l0,
    const float* __restrict__ w_l1, const float* __restrict__ b_l1,
    const float* __restrict__ a_l1,
    const float* __restrict__ w_l2, const float* __restrict__ b_l2,
    const float* __restrict__ a_l2,
    const float* __restrict__ wl, const float* __restrict__ bl,
    float* __restrict__ dos /* [NGRAPH][NE] */) {
  __shared__ __align__(32) h16 sA[2][2 * CH];    // double-buffered A chunk
  __shared__ __align__(32) h16 sB[2][16 * CH];   // 32KB
  __shared__ __align__(32) h16 sH[2 * 8 * CH];   // frag-major activations, 16KB
  __shared__ float pS[32 * 4], pQ[32 * 4];
  __shared__ float sMu[32], sRinv[32];

  const int tid  = threadIdx.x;
  const int lane = tid & 31;
  const int wv   = tid >> 5;
  const int mi   = wv >> 2;
  const int nj   = wv & 3;
  const int hic  = (lane >> 4) & 1;
  const int nn   = lane & 15;
  const int row0 = blockIdx.x * 32;

  // A-chunk loader for feat = [emb | graph]
  auto load_a_feat = [&](int k0) -> v16h {
    int mtile = tid >> 5, slane = tid & 31;
    int row = row0 + mtile * 16 + (slane & 15);
    int e10 = row >> 10, gg = row & 1023;
    int ahi = slane >> 4;
    v16h v;
#pragma unroll
    for (int e = 0; e < 16; ++e) {
      int k = k0 + (e & 7) + 8 * ahi + 16 * (e >> 3);
      float f = (k < HD) ? emb[(size_t)e10 * HD + k]
                         : graph[(size_t)gg * HD + (k - HD)];
      v[e] = (h16)f;
    }
    return v;
  };

  // ---- GEMM0 over K=512, double-buffered ----
  v8f acc[4];
#pragma unroll
  for (int t = 0; t < 4; ++t) acc[t] = zero8();

  BReg pb = load_b(w0, 0, 2 * HD, tid);
  v16h pa = zeroh();
  if (tid < 64) pa = load_a_feat(0);
  store_b(sB[0], tid, pb);
  if (tid < 64) *(v16h*)(sA[0] + tid * 16) = pa;
  __syncthreads();

  for (int kc = 0; kc < 16; ++kc) {
    const h16* curB = sB[kc & 1];
    const h16* curA = sA[kc & 1];
    // prefetch: next w0 chunk, or layer-0 chunk 0 on the last iteration
    pb = (kc < 15) ? load_b(w0, (kc + 1) * 32, 2 * HD, tid)
                   : load_b(w_l0, 0, HD, tid);
    if (kc < 15 && tid < 64) pa = load_a_feat((kc + 1) * 32);
    v16h a = read_frag(curA + mi * CH, lane);
#pragma unroll
    for (int t = 0; t < 4; ++t) {
      v16h b = read_frag(curB + (nj * 4 + t) * CH, lane);
      acc[t] = WMMA_F16(a, b, acc[t]);
    }
    store_b(sB[(kc + 1) & 1], tid, pb);   // (15+1)&1 == 0: layer-0 chunk in sB[0]
    if (kc < 15 && tid < 64) *(v16h*)(sA[(kc + 1) & 1] + tid * 16) = pa;
    __syncthreads();
  }

  // bias
#pragma unroll
  for (int t = 0; t < 4; ++t) {
    float bb = b0[nj * 64 + t * 16 + nn];
#pragma unroll
    for (int vv = 0; vv < 8; ++vv) acc[t][vv] += bb;
  }

  // ---- LayerNorm: per-row mean/var via in-wave + cross-wave reduction ----
  float rs[8], rq[8];
#pragma unroll
  for (int vv = 0; vv < 8; ++vv) {
    float s = 0.f, q = 0.f;
#pragma unroll
    for (int t = 0; t < 4; ++t) {
      float xv = acc[t][vv];
      s += xv;
      q += xv * xv;
    }
    rs[vv] = s;
    rq[vv] = q;
  }
#pragma unroll
  for (int off = 1; off < 16; off <<= 1) {
#pragma unroll
    for (int vv = 0; vv < 8; ++vv) {
      rs[vv] += __shfl_xor(rs[vv], off, 32);
      rq[vv] += __shfl_xor(rq[vv], off, 32);
    }
  }
  if (nn == 0) {
#pragma unroll
    for (int vv = 0; vv < 8; ++vv) {
      int m = mi * 16 + vv + (hic ? 8 : 0);
      pS[m * 4 + nj] = rs[vv];
      pQ[m * 4 + nj] = rq[vv];
    }
  }
  __syncthreads();
  if (tid < 32) {
    float s = pS[tid * 4] + pS[tid * 4 + 1] + pS[tid * 4 + 2] + pS[tid * 4 + 3];
    float q = pQ[tid * 4] + pQ[tid * 4 + 1] + pQ[tid * 4 + 2] + pQ[tid * 4 + 3];
    float mu = s * (1.f / HD);
    float var = q * (1.f / HD) - mu * mu;
    sMu[tid] = mu;
    sRinv[tid] = rsqrtf(var + LN_EPS);
  }
  __syncthreads();

  const float a0 = pa0[0];
#pragma unroll
  for (int t = 0; t < 4; ++t) {
    int c = nj * 64 + t * 16 + nn;
    float gsc = lng[c], bsc = lnb[c];
    int kcA  = nj * 2 + (t >> 1);
    int elem = (nn & 7) + 8 * (t & 1);
#pragma unroll
    for (int vv = 0; vv < 8; ++vv) {
      int m = mi * 16 + vv + (hic ? 8 : 0);
      float xv = (acc[t][vv] - sMu[m]) * sRinv[m] * gsc + bsc;
      xv = (xv >= 0.f) ? xv : a0 * xv;
      sH[((mi * 8 + kcA) * 32 + (m & 15) + 16 * (nn >> 3)) * 16 + elem] = (h16)xv;
    }
  }
  __syncthreads();  // sH visible before layer-0 reads (chunk0 already in sB[0])

  // ---- 3 hidden layers, activations resident in LDS (fragment-major) ----
  const float* lw[3] = {w_l0, w_l1, w_l2};
  const float* lb[3] = {b_l0, b_l1, b_l2};
  const float* la[3] = {a_l0, a_l1, a_l2};
  for (int L = 0; L < 3; ++L) {
    v8f acc2[4];
#pragma unroll
    for (int t = 0; t < 4; ++t) acc2[t] = zero8();
    for (int kc = 0; kc < 8; ++kc) {
      const h16* cur = sB[kc & 1];
      bool pf = (kc < 7) || (L < 2);
      if (kc < 7)      pb = load_b(lw[L], (kc + 1) * 32, HD, tid);
      else if (L < 2)  pb = load_b(lw[L + 1], 0, HD, tid);
      v16h a = read_frag(sH + (mi * 8 + kc) * CH, lane);
#pragma unroll
      for (int t = 0; t < 4; ++t) {
        v16h b = read_frag(cur + (nj * 4 + t) * CH, lane);
        acc2[t] = WMMA_F16(a, b, acc2[t]);
      }
      if (pf) store_b(sB[(kc + 1) & 1], tid, pb);  // (7+1)&1==0: next layer in sB[0]
      __syncthreads();
    }
    float aL = la[L][0];
#pragma unroll
    for (int t = 0; t < 4; ++t) {
      int c = nj * 64 + t * 16 + nn;
      float bb = lb[L][c];
      int kcA  = nj * 2 + (t >> 1);
      int elem = (nn & 7) + 8 * (t & 1);
#pragma unroll
      for (int vv = 0; vv < 8; ++vv) {
        int m15 = vv + 8 * hic;
        float xv = acc2[t][vv] + bb;
        xv = (xv >= 0.f) ? xv : aL * xv;
        sH[((mi * 8 + kcA) * 32 + m15 + 16 * (nn >> 3)) * 16 + elem] = (h16)xv;
      }
    }
    __syncthreads();  // sH rewrite visible before next layer / final dot
  }

  // ---- final projection H -> 1, transpose into dos[g][e] ----
  if (tid < 32) {
    int row = row0 + tid;
    int e10 = row >> 10, gg = row & 1023;
    int mtile = tid >> 4, m15 = tid & 15;
    float s = bl[0];
    for (int c = 0; c < HD; ++c) {
      int kc = c >> 5, kk = c & 31;
      int ahi = (kk >> 3) & 1;
      int elem = (kk & 7) | (((kk >> 4) & 1) << 3);
      float hv = (float)sH[((mtile * 8 + kc) * 32 + m15 + 16 * ahi) * 16 + elem];
      s += hv * wl[c];
    }
    dos[(size_t)gg * NE + e10] = s;
  }
}

// ---------------------------------------------------------------------------
// Input order: setup_inputs() dict order, params flattened in insertion order:
// 0:x 1:z 2:pos 3:batch 4:edge_index 5:emb 6:node_w1 7:node_b1 8:node_a
// 9:node_w2 10:node_b2 11:edge_w1 12:edge_b1 13:edge_a 14:edge_w2 15:edge_b2
// 16:dec_w 17:dec_b 18:out_w0 19:out_b0 20:ln_g 21:ln_b 22:out_a0
// 23..31: layers[i].{w,b,a} 32:out_wl 33:out_bl
// pos / edge_* are dead in the reference (ea is deleted) and are not read.
extern "C" void kernel_launch(void* const* d_in, const int* in_sizes, int n_in,
                              void* d_out, int out_size, void* d_ws,
                              size_t ws_size, hipStream_t stream) {
  (void)in_sizes; (void)n_in; (void)out_size; (void)ws_size;
  const float* x       = (const float*)d_in[0];
  const float* z       = (const float*)d_in[1];
  const int*   batch   = (const int*)d_in[3];
  const float* emb     = (const float*)d_in[5];
  const float* node_w1 = (const float*)d_in[6];
  const float* node_b1 = (const float*)d_in[7];
  const float* node_a  = (const float*)d_in[8];
  const float* node_w2 = (const float*)d_in[9];
  const float* node_b2 = (const float*)d_in[10];
  const float* dec_w   = (const float*)d_in[16];
  const float* dec_b   = (const float*)d_in[17];
  const float* out_w0  = (const float*)d_in[18];
  const float* out_b0  = (const float*)d_in[19];
  const float* ln_g    = (const float*)d_in[20];
  const float* ln_b    = (const float*)d_in[21];
  const float* out_a0  = (const float*)d_in[22];
  const float* l0w = (const float*)d_in[23];
  const float* l0b = (const float*)d_in[24];
  const float* l0a = (const float*)d_in[25];
  const float* l1w = (const float*)d_in[26];
  const float* l1b = (const float*)d_in[27];
  const float* l1a = (const float*)d_in[28];
  const float* l2w = (const float*)d_in[29];
  const float* l2b = (const float*)d_in[30];
  const float* l2a = (const float*)d_in[31];
  const float* out_wl = (const float*)d_in[32];
  const float* out_bl = (const float*)d_in[33];
  float* dos = (float*)d_out;

  float* pooled = (float*)d_ws;                      // [NGRAPH][512] f32, 2 MB
  float* graphb = pooled + (size_t)NGRAPH * 2 * HD;  // [NGRAPH][256] f32, 1 MB

  k_zero<<<512, 256, 0, stream>>>(pooled, NGRAPH * 2 * HD);

  dim3 g1(NNODES / 32, 2);
  k_node_pool<<<g1, 256, 0, stream>>>(x, z, batch, node_w1, node_b1, node_a,
                                      node_w2, node_b2, pooled);

  k_decoder<<<NGRAPH / 32, 256, 0, stream>>>(pooled, dec_w, dec_b, graphb);

  k_head<<<(NE * NGRAPH) / 32, 256, 0, stream>>>(
      emb, graphb, out_w0, out_b0, ln_g, ln_b, out_a0, l0w, l0b, l0a, l1w, l1b,
      l1a, l2w, l2b, l2a, out_wl, out_bl, dos);
}
